// GATTransformer_78958678769914
// MI455X (gfx1250) — matrix operand order repaired and checked by reading, not compiled, not run on hardware
//
#include <hip/hip_runtime.h>
#include <hip/hip_bf16.h>
#include <math.h>

#define N_NODES 50000
#define N_FRAMES 5
#define N_EDGES 400000
#define E_TOT (N_EDGES + N_NODES)
#define H1 8
#define C1 8
#define D_MODEL 64
#define N_HEADS 4
#define HEAD_DIM 16
#define FF_DIM 128
#define N_TLAYERS 3
#define LN_EPS 1e-5f

typedef __attribute__((ext_vector_type(16))) _Float16 v16h;
typedef __attribute__((ext_vector_type(8)))  _Float16 v8h;
typedef __attribute__((ext_vector_type(8)))  float    v8f;

// ---------------------------------------------------------------------------
// Pack f32 (rows x K) -> f16 (rows x KP), zero-padded in K.
// ---------------------------------------------------------------------------
__global__ void pack_f16_kernel(const float* __restrict__ src, int K,
                                _Float16* __restrict__ dst, int Kp, int rows)
{
    int tid = blockIdx.x * blockDim.x + threadIdx.x;
    if (tid >= rows * Kp) return;
    int r = tid / Kp, k = tid - r * Kp;
    float v = (k < K) ? src[(size_t)r * K + k] : 0.f;
    dst[tid] = (_Float16)v;
}

// ---------------------------------------------------------------------------
// WMMA GEMM on packed f16:  C(N x 64) = A(N x KP f16) @ W^T  (W: 64 x KP f16)
// One wave -> one 16-row stripe (4 x 16x16 tiles). A fragment loaded once per
// k-step and reused across the 4 column tiles. Fully unrolled (KP template).
// ---------------------------------------------------------------------------
template <int KP>
__global__ void wmma_gemm_packed(const _Float16* __restrict__ A,
                                 const _Float16* __restrict__ W,
                                 float* __restrict__ C, int N)
{
    int wave = (blockIdx.x * blockDim.x + threadIdx.x) >> 5;
    int lane = threadIdx.x & 31;
    int tiles_m = N >> 4;                 // N is a multiple of 16
    if (wave >= tiles_m) return;          // wave-uniform

    int l15 = lane & 15;
    int hi  = lane >> 4;                  // 0: lanes 0-15, 1: lanes 16-31
    int arow = wave * 16 + l15;

    v8f acc[4];
#pragma unroll
    for (int tn = 0; tn < 4; ++tn) acc[tn] = v8f{0.f,0.f,0.f,0.f,0.f,0.f,0.f,0.f};

#pragma unroll
    for (int k0 = 0; k0 < KP; k0 += 32) {
        // A fragment: halves [kb..kb+7] and [kb+16..kb+23]  (kb = k0 + hi*8)
        const _Float16* abase = A + (size_t)arow * KP + k0 + hi * 8;
        v8h a0 = *(const v8h*)(abase);
        v8h a1 = *(const v8h*)(abase + 16);
        v16h a;
#pragma unroll
        for (int i = 0; i < 8; ++i) { a[i] = a0[i]; a[8 + i] = a1[i]; }

#pragma unroll
        for (int tn = 0; tn < 4; ++tn) {
            // B fragment: column (tn*16+l15) = W row; halves [kb..kb+7] and [kb+16..kb+23]  (kb = k0 + hi*8)
            const _Float16* bbase = W + (size_t)(tn * 16 + l15) * KP + k0 + hi * 8;
            v8h b0 = *(const v8h*)(bbase);
            v8h b1 = *(const v8h*)(bbase + 16);
            v16h b;
#pragma unroll
            for (int i = 0; i < 8; ++i) { b[i] = b0[i]; b[8 + i] = b1[i]; }

            acc[tn] = __builtin_amdgcn_wmma_f32_16x16x32_f16(
                          false, a, false, b, (short)0, acc[tn], false, false);
        }
    }

#pragma unroll
    for (int tn = 0; tn < 4; ++tn) {
#pragma unroll
        for (int i = 0; i < 8; ++i) {
            int r = wave * 16 + hi * 8 + i;   // D VGPR i: M = hi*8+i
            C[(size_t)r * 64 + tn * 16 + l15] = acc[tn][i];
        }
    }
}

// ---------------------------------------------------------------------------
// Monotone-uint encoding for float atomic max
// ---------------------------------------------------------------------------
__device__ __forceinline__ unsigned fenc(float f) {
    unsigned u = __float_as_uint(f);
    return (u & 0x80000000u) ? ~u : (u | 0x80000000u);
}
__device__ __forceinline__ float fdec(unsigned m) {
    unsigned u = (m & 0x80000000u) ? (m ^ 0x80000000u) : ~m;
    return __uint_as_float(u);
}

// a_s[n,h] = sum_c h[n,h*ch+c]*att_src[h,c] ; same for a_d.  heads*ch == 64
__global__ void attn_coef_kernel(const float* __restrict__ h,
                                 const float* __restrict__ asrc,
                                 const float* __restrict__ adst,
                                 float* __restrict__ as_o,
                                 float* __restrict__ ad_o,
                                 int heads, int ch)
{
    int tid = blockIdx.x * blockDim.x + threadIdx.x;
    if (tid >= N_NODES * heads) return;
    int n = tid / heads, hd = tid % heads;
    float s = 0.f, d = 0.f;
    for (int c = 0; c < ch; ++c) {
        float v = h[(size_t)n * 64 + hd * ch + c];
        s += v * asrc[hd * ch + c];
        d += v * adst[hd * ch + c];
    }
    as_o[tid] = s;
    ad_o[tid] = d;
}

__global__ void gat_init_kernel(unsigned* __restrict__ emax,
                                float* __restrict__ denom,
                                float* __restrict__ out, int heads)
{
    int tid = blockIdx.x * blockDim.x + threadIdx.x;
    if (tid < N_NODES * heads) {
        emax[tid]  = fenc(-__builtin_inff());
        denom[tid] = 0.f;
    }
    if (tid < N_NODES * 64) out[tid] = 0.f;
}

__device__ __forceinline__ void edge_sd(const int* __restrict__ srcI,
                                        const int* __restrict__ dstI,
                                        int e, int& s, int& d)
{
    if (e < N_EDGES) { s = srcI[e]; d = dstI[e]; }
    else             { s = d = e - N_EDGES; }       // self loops
}

__device__ __forceinline__ float lrelu02(float x) {
    return (x < 0.f) ? 0.2f * x : x;
}

__global__ void edge_max_kernel(const int* __restrict__ srcI, const int* __restrict__ dstI,
                                const float* __restrict__ as_b, const float* __restrict__ ad_b,
                                unsigned* __restrict__ emax, int heads)
{
    int tid = blockIdx.x * blockDim.x + threadIdx.x;
    if (tid >= E_TOT * heads) return;
    int e = tid / heads, hd = tid % heads;
    int s, d; edge_sd(srcI, dstI, e, s, d);
    float ev = lrelu02(as_b[(size_t)s * heads + hd] + ad_b[(size_t)d * heads + hd]);
    atomicMax(&emax[(size_t)d * heads + hd], fenc(ev));
}

__global__ void edge_sum_kernel(const int* __restrict__ srcI, const int* __restrict__ dstI,
                                const float* __restrict__ as_b, const float* __restrict__ ad_b,
                                const unsigned* __restrict__ emax,
                                float* __restrict__ denom, int heads)
{
    int tid = blockIdx.x * blockDim.x + threadIdx.x;
    if (tid >= E_TOT * heads) return;
    int e = tid / heads, hd = tid % heads;
    int s, d; edge_sd(srcI, dstI, e, s, d);
    float ev = lrelu02(as_b[(size_t)s * heads + hd] + ad_b[(size_t)d * heads + hd]);
    float ex = __expf(ev - fdec(emax[(size_t)d * heads + hd]));
    atomicAdd(&denom[(size_t)d * heads + hd], ex);
}

__global__ void edge_acc_kernel(const int* __restrict__ srcI, const int* __restrict__ dstI,
                                const float* __restrict__ as_b, const float* __restrict__ ad_b,
                                const unsigned* __restrict__ emax,
                                const float* __restrict__ denom,
                                const float* __restrict__ feat,   // N x 64
                                float* __restrict__ out,          // N x 64 (accumulate)
                                int heads, int ch)
{
    int tid = blockIdx.x * blockDim.x + threadIdx.x;
    if (tid >= E_TOT * heads) return;
    int e = tid / heads, hd = tid % heads;
    int s, d; edge_sd(srcI, dstI, e, s, d);
    float ev = lrelu02(as_b[(size_t)s * heads + hd] + ad_b[(size_t)d * heads + hd]);
    float ex = __expf(ev - fdec(emax[(size_t)d * heads + hd]));
    float alpha = ex / denom[(size_t)d * heads + hd];
    for (int c = 0; c < ch; ++c) {
        atomicAdd(&out[(size_t)d * 64 + hd * ch + c],
                  feat[(size_t)s * 64 + hd * ch + c] * alpha);
    }
}

// h16 = (f16) elu(in + b)  over N x 64 -- feeds only the layer-2 WMMA GEMM
__global__ void bias_elu_f16_kernel(const float* __restrict__ in,
                                    const float* __restrict__ b,
                                    _Float16* __restrict__ out16)
{
    int tid = blockIdx.x * blockDim.x + threadIdx.x;
    if (tid >= N_NODES * 64) return;
    float v = in[tid] + b[tid & 63];
    v = (v > 0.f) ? v : (__expf(v) - 1.f);
    out16[tid] = (_Float16)v;
}

// emb[j] = mean_n out[n,j] + b2[j]   (one block per feature)
__global__ void col_mean_kernel(const float* __restrict__ out2,
                                const float* __restrict__ b2,
                                float* __restrict__ emb_f)
{
    __shared__ float sm[256];
    int j = blockIdx.x;
    float s = 0.f;
    for (int n = threadIdx.x; n < N_NODES; n += blockDim.x)
        s += out2[(size_t)n * 64 + j];
    sm[threadIdx.x] = s;
    __syncthreads();
    for (int off = 128; off > 0; off >>= 1) {
        if ((int)threadIdx.x < off) sm[threadIdx.x] += sm[threadIdx.x + off];
        __syncthreads();
    }
    if (threadIdx.x == 0)
        emb_f[j] = sm[0] * (1.f / (float)N_NODES) + b2[j];
}

// ---------------------------------------------------------------------------
// Transformer (3 layers over 5x64) + MLP head, one block of 128 threads.
// ---------------------------------------------------------------------------
__device__ __forceinline__ void ln_rows(const float* __restrict__ in, int istride,
                                        float* __restrict__ outp, int ostride,
                                        const float* __restrict__ w,
                                        const float* __restrict__ b,
                                        float* red, int j)
{
    for (int t = 0; t < N_FRAMES; ++t) {
        const float* x = in + t * istride;
        red[j] = (j < 64) ? x[j] : 0.f;
        __syncthreads();
        for (int off = 64; off > 0; off >>= 1) {
            if (j < off) red[j] += red[j + off];
            __syncthreads();
        }
        float mean = red[0] * (1.f / 64.f);
        __syncthreads();
        float dv = (j < 64) ? (x[j] - mean) : 0.f;
        red[j] = dv * dv;
        __syncthreads();
        for (int off = 64; off > 0; off >>= 1) {
            if (j < off) red[j] += red[j + off];
            __syncthreads();
        }
        float var = red[0] * (1.f / 64.f);
        __syncthreads();
        if (j < 64)
            outp[t * ostride + j] = (x[j] - mean) * rsqrtf(var + LN_EPS) * w[j] + b[j];
        __syncthreads();
    }
}

__global__ void transformer_head_kernel(
    const float* __restrict__ emb,      // 5 x 64
    const float* __restrict__ Wqkv, const float* __restrict__ bqkv,
    const float* __restrict__ Wo,   const float* __restrict__ bo,
    const float* __restrict__ ln1w, const float* __restrict__ ln1b,
    const float* __restrict__ Wff1, const float* __restrict__ bff1,
    const float* __restrict__ Wff2, const float* __restrict__ bff2,
    const float* __restrict__ ln2w, const float* __restrict__ ln2b,
    const float* __restrict__ Wh1,  const float* __restrict__ bh1,
    const float* __restrict__ Wh2,  const float* __restrict__ bh2,
    float* __restrict__ out)
{
    __shared__ float seq[N_FRAMES][64];
    __shared__ float qkv[N_FRAMES][192];
    __shared__ float att[N_HEADS][N_FRAMES][N_FRAMES];
    __shared__ float tmp[N_FRAMES][64];
    __shared__ float ffh[N_FRAMES][FF_DIM];
    __shared__ float red[128];
    __shared__ float hid[32];

    int j = threadIdx.x;
    for (int i = j; i < N_FRAMES * 64; i += blockDim.x)
        seq[i / 64][i % 64] = emb[i];
    __syncthreads();

    const float scale = 0.25f; // 1/sqrt(HEAD_DIM=16)

    for (int l = 0; l < N_TLAYERS; ++l) {
        const float* Wq = Wqkv + (size_t)l * 192 * 64;
        const float* bq = bqkv + (size_t)l * 192;
        for (int oi = j; oi < N_FRAMES * 192; oi += blockDim.x) {
            int t = oi / 192, o = oi % 192;
            float s = bq[o];
            for (int k = 0; k < 64; ++k) s += seq[t][k] * Wq[o * 64 + k];
            qkv[t][o] = s;
        }
        __syncthreads();
        if (j < N_HEADS * N_FRAMES) {
            int h = j / N_FRAMES, tq = j % N_FRAMES;
            float sc[N_FRAMES];
            float mx = -1e30f;
            for (int tk = 0; tk < N_FRAMES; ++tk) {
                float s = 0.f;
                for (int dd = 0; dd < HEAD_DIM; ++dd)
                    s += qkv[tq][h * HEAD_DIM + dd] * qkv[tk][64 + h * HEAD_DIM + dd];
                sc[tk] = s * scale;
                mx = fmaxf(mx, sc[tk]);
            }
            float den = 0.f;
            for (int tk = 0; tk < N_FRAMES; ++tk) { sc[tk] = __expf(sc[tk] - mx); den += sc[tk]; }
            for (int tk = 0; tk < N_FRAMES; ++tk) att[h][tq][tk] = sc[tk] / den;
        }
        __syncthreads();
        if (j < 64) {
            int h = j / HEAD_DIM, dd = j % HEAD_DIM;
            for (int t = 0; t < N_FRAMES; ++t) {
                float s = 0.f;
                for (int tk = 0; tk < N_FRAMES; ++tk)
                    s += att[h][t][tk] * qkv[tk][128 + h * HEAD_DIM + dd];
                tmp[t][j] = s;
            }
        }
        __syncthreads();
        const float* Wol = Wo + (size_t)l * 64 * 64;
        const float* bol = bo + (size_t)l * 64;
        if (j < 64) {
            for (int t = 0; t < N_FRAMES; ++t) {
                float s = bol[j];
                for (int k = 0; k < 64; ++k) s += tmp[t][k] * Wol[j * 64 + k];
                qkv[t][j] = seq[t][j] + s;
            }
        }
        __syncthreads();
        ln_rows(&qkv[0][0], 192, &seq[0][0], 64, ln1w + l * 64, ln1b + l * 64, red, j);
        const float* W1l = Wff1 + (size_t)l * FF_DIM * 64;
        const float* b1l = bff1 + (size_t)l * FF_DIM;
        for (int oi = j; oi < N_FRAMES * FF_DIM; oi += blockDim.x) {
            int t = oi / FF_DIM, i = oi % FF_DIM;
            float s = b1l[i];
            for (int k = 0; k < 64; ++k) s += seq[t][k] * W1l[i * 64 + k];
            ffh[t][i] = (s > 0.f) ? s : 0.f;
        }
        __syncthreads();
        const float* W2l = Wff2 + (size_t)l * 64 * FF_DIM;
        const float* b2l = bff2 + (size_t)l * 64;
        if (j < 64) {
            for (int t = 0; t < N_FRAMES; ++t) {
                float s = b2l[j];
                for (int k = 0; k < FF_DIM; ++k) s += ffh[t][k] * W2l[j * FF_DIM + k];
                qkv[t][j] = seq[t][j] + s;
            }
        }
        __syncthreads();
        ln_rows(&qkv[0][0], 192, &seq[0][0], 64, ln2w + l * 64, ln2b + l * 64, red, j);
    }

    if (j < 32) {
        float s = bh1[j];
        for (int k = 0; k < 64; ++k) s += seq[N_FRAMES - 1][k] * Wh1[j * 64 + k];
        hid[j] = (s > 0.f) ? s : 0.f;
    }
    __syncthreads();
    if (j < 2) {
        float s = bh2[j];
        for (int k = 0; k < 32; ++k) s += hid[k] * Wh2[j * 32 + k];
        out[j] = s;
    }
}

// ---------------------------------------------------------------------------
// Launch
// ---------------------------------------------------------------------------
extern "C" void kernel_launch(void* const* d_in, const int* in_sizes, int n_in,
                              void* d_out, int out_size, void* d_ws, size_t ws_size,
                              hipStream_t stream)
{
    (void)in_sizes; (void)n_in; (void)out_size; (void)ws_size;

    const float* x         = (const float*)d_in[0];
    const int*   edge_idx  = (const int*)  d_in[1];
    const float* W1        = (const float*)d_in[2];
    const float* att_src1  = (const float*)d_in[3];
    const float* att_dst1  = (const float*)d_in[4];
    const float* b1        = (const float*)d_in[5];
    const float* W2        = (const float*)d_in[6];
    const float* att_src2  = (const float*)d_in[7];
    const float* att_dst2  = (const float*)d_in[8];
    const float* b2        = (const float*)d_in[9];
    const float* tWqkv     = (const float*)d_in[10];
    const float* tbqkv     = (const float*)d_in[11];
    const float* tWo       = (const float*)d_in[12];
    const float* tbo       = (const float*)d_in[13];
    const float* tln1w     = (const float*)d_in[14];
    const float* tln1b     = (const float*)d_in[15];
    const float* tWff1     = (const float*)d_in[16];
    const float* tbff1     = (const float*)d_in[17];
    const float* tWff2     = (const float*)d_in[18];
    const float* tbff2     = (const float*)d_in[19];
    const float* tln2w     = (const float*)d_in[20];
    const float* tln2b     = (const float*)d_in[21];
    const float* Wh1       = (const float*)d_in[22];
    const float* bh1       = (const float*)d_in[23];
    const float* Wh2       = (const float*)d_in[24];
    const float* bh2       = (const float*)d_in[25];

    float* ws = (float*)d_ws;
    size_t off = 0;
    float*    bufA    = ws + off; off += (size_t)N_NODES * 64;     // GEMM output (lin features)
    float*    bufB    = ws + off; off += (size_t)N_NODES * 64;     // GAT accumulator
    float*    as_b    = ws + off; off += (size_t)N_NODES * H1;
    float*    ad_b    = ws + off; off += (size_t)N_NODES * H1;
    unsigned* emax_b  = (unsigned*)(ws + off); off += (size_t)N_NODES * H1;
    float*    denom_b = ws + off; off += (size_t)N_NODES * H1;
    float*    emb     = ws + off; off += (size_t)N_FRAMES * 64;
    _Float16* packX   = (_Float16*)(ws + off); off += (size_t)N_NODES * 32 / 2;  // N x 32 f16
    _Float16* packH   = (_Float16*)(ws + off); off += (size_t)N_NODES * 64 / 2;  // N x 64 f16
    _Float16* packW1  = (_Float16*)(ws + off); off += 64 * 32 / 2;               // 64 x 32 f16
    _Float16* packW2  = (_Float16*)(ws + off); off += 64 * 64 / 2;               // 64 x 64 f16

    const int TB = 256;
    const int gemm_waves  = N_NODES / 16;                        // one wave per 16-row stripe
    const int gemm_blocks = (gemm_waves * 32 + TB - 1) / TB;
    const int n64_blocks  = (N_NODES * 64 + TB - 1) / TB;

    // Pack weights once (f32 -> f16, zero-padded K)
    pack_f16_kernel<<<(64 * 32 + TB - 1) / TB, TB, 0, stream>>>(W1, 13, packW1, 32, 64);
    pack_f16_kernel<<<(64 * 64 + TB - 1) / TB, TB, 0, stream>>>(W2, 64, packW2, 64, 64);

    for (int f = 0; f < N_FRAMES; ++f) {
        const float* xf   = x + (size_t)f * N_NODES * 13;
        const int*   srcI = edge_idx + (size_t)f * 2 * N_EDGES;
        const int*   dstI = srcI + N_EDGES;

        // ---- GAT layer 1 (heads=8, ch=8) ----
        pack_f16_kernel<<<(N_NODES * 32 + TB - 1) / TB, TB, 0, stream>>>(xf, 13, packX, 32, N_NODES);
        wmma_gemm_packed<32><<<gemm_blocks, TB, 0, stream>>>(packX, packW1, bufA, N_NODES);
        attn_coef_kernel<<<(N_NODES * H1 + TB - 1) / TB, TB, 0, stream>>>(
            bufA, att_src1, att_dst1, as_b, ad_b, H1, C1);
        gat_init_kernel<<<n64_blocks, TB, 0, stream>>>(emax_b, denom_b, bufB, H1);
        int eth = E_TOT * H1;
        edge_max_kernel<<<(eth + TB - 1) / TB, TB, 0, stream>>>(srcI, dstI, as_b, ad_b, emax_b, H1);
        edge_sum_kernel<<<(eth + TB - 1) / TB, TB, 0, stream>>>(srcI, dstI, as_b, ad_b, emax_b, denom_b, H1);
        edge_acc_kernel<<<(eth + TB - 1) / TB, TB, 0, stream>>>(srcI, dstI, as_b, ad_b, emax_b, denom_b,
                                                                bufA, bufB, H1, C1);
        bias_elu_f16_kernel<<<n64_blocks, TB, 0, stream>>>(bufB, b1, packH);

        // ---- GAT layer 2 (heads=1, ch=64) ----
        wmma_gemm_packed<64><<<gemm_blocks, TB, 0, stream>>>(packH, packW2, bufA, N_NODES);
        attn_coef_kernel<<<(N_NODES + TB - 1) / TB, TB, 0, stream>>>(
            bufA, att_src2, att_dst2, as_b, ad_b, 1, 64);
        gat_init_kernel<<<n64_blocks, TB, 0, stream>>>(emax_b, denom_b, bufB, 1);
        edge_max_kernel<<<(E_TOT + TB - 1) / TB, TB, 0, stream>>>(srcI, dstI, as_b, ad_b, emax_b, 1);
        edge_sum_kernel<<<(E_TOT + TB - 1) / TB, TB, 0, stream>>>(srcI, dstI, as_b, ad_b, emax_b, denom_b, 1);
        edge_acc_kernel<<<(E_TOT + TB - 1) / TB, TB, 0, stream>>>(srcI, dstI, as_b, ad_b, emax_b, denom_b,
                                                                  bufA, bufB, 1, 64);
        col_mean_kernel<<<64, 256, 0, stream>>>(bufB, b2, emb + (size_t)f * 64);
    }

    transformer_head_kernel<<<1, 128, 0, stream>>>(
        emb, tWqkv, tbqkv, tWo, tbo, tln1w, tln1b,
        tWff1, tbff1, tWff2, tbff2, tln2w, tln2b,
        Wh1, bh1, Wh2, bh2, (float*)d_out);
}
